// T_Painter_7559142441594
// MI455X (gfx1250) — matrix operand order repaired and checked by reading, not compiled, not run on hardware
//
#include <hip/hip_runtime.h>
#include <hip/hip_bf16.h>

// ---------------------------------------------------------------------------
// ViTDet block for MI455X (gfx1250, wave32, WMMA).
// All heavy math runs through v_wmma_f32_16x16x32_bf16 (bf16 in, f32 acc).
// ---------------------------------------------------------------------------

#define BB   8
#define HH_  28
#define WW_  28
#define NN   784          // H*W
#define CC   1024
#define NHEADS 16
#define HD   64
#define HID_ 4096
#define BC_  512
#define MM   (BB*NN)      // 6272 = 49*128
#define NPAD 896          // keys padded to 896 = 56*16 (PV half = 448 = 14*32)
#define SCALE_ 0.125f     // HD^-0.5
#define EPS_LN  1e-5f
#define EPS_CLN 1e-6f

typedef __attribute__((ext_vector_type(16))) __bf16 v16bf;
typedef __attribute__((ext_vector_type(8)))  float  v8f;
typedef __attribute__((ext_vector_type(4)))  float  f32x4;

union FragU { v16bf v; f32x4 q[2]; };

// ---- WMMA fragment loaders -------------------------------------------------
// A-operand (16x32, bf16, row-major, row = lane%16):
//   lanes 0-15 hold K 0..7 & 16..23; lanes 16-31 hold K 8..15 & 24..31
__device__ inline v16bf load_fragA(const __bf16* base /* row start + kk */, int lane) {
  FragU u;
  const int kb = (lane & 16) ? 8 : 0;
  u.q[0] = *reinterpret_cast<const f32x4*>(base + kb);
  u.q[1] = *reinterpret_cast<const f32x4*>(base + kb + 16);
  return u.v;
}
__device__ inline v16bf load_fragA_guard(const __bf16* base, int lane, bool valid) {
  FragU u;
  if (valid) {
    const int kb = (lane & 16) ? 8 : 0;
    u.q[0] = *reinterpret_cast<const f32x4*>(base + kb);
    u.q[1] = *reinterpret_cast<const f32x4*>(base + kb + 16);
  } else {
    f32x4 z = {0.f, 0.f, 0.f, 0.f};
    u.q[0] = z; u.q[1] = z;
  }
  return u.v;
}
// B-operand (32x16, bf16), column = lane%16, sourced from Wt[col][K] row-major:
//   lanes 0-15 hold K 0..7 & 16..23; lanes 16-31 hold K 8..15 & 24..31
__device__ inline v16bf load_fragB(const __bf16* btcol /* Wt + col*K + kk */, int lane) {
  FragU u;
  const int kb = (lane & 16) ? 8 : 0;
  const f32x4* p = reinterpret_cast<const f32x4*>(btcol + kb);
  u.q[0] = p[0];
  u.q[1] = p[2];
  return u.v;
}

__device__ inline float gelu_f(float x) {
  return 0.5f * x * (1.0f + erff(x * 0.70710678118f));
}

// ---- shared GEMM core: one wave -> 16x64 tile ------------------------------
__device__ inline void gemm_core4(const __bf16* Abase, const __bf16* Bt, int K,
                                  int c0, int lane, v8f acc[4]) {
  const int colb = lane & 15;
  for (int kk = 0; kk < K; kk += 32) {
    v16bf a = load_fragA(Abase + kk, lane);
#pragma unroll
    for (int t = 0; t < 4; ++t) {
      v16bf b = load_fragB(Bt + (size_t)(c0 + t * 16 + colb) * K + kk, lane);
      acc[t] = __builtin_amdgcn_wmma_f32_16x16x32_bf16(
          false, a, false, b, (short)0, acc[t], false, false);
    }
  }
}

// ---- generic GEMM: out[M][Nc] = A[M][K] @ Wt[Nc][K]^T ----------------------
// mode 0: (+bias) -> bf16   | mode 1: (+bias, GELU) -> bf16
// mode 2: (+bias, +res) -> f32 | mode 3: plain -> f32
__global__ __launch_bounds__(256)
void gemm_ws_kernel(const __bf16* __restrict__ A, const __bf16* __restrict__ Bt,
                    const float* __restrict__ bias, const float* __restrict__ res,
                    float* __restrict__ outf, __bf16* __restrict__ outb,
                    int Nc, int K, int mode) {
  const int lane = threadIdx.x & 31, wv = threadIdx.x >> 5;
  const int r0 = blockIdx.x * 128 + wv * 16;
  const int c0 = blockIdx.y * 64;
  const __bf16* Abase = A + (size_t)(r0 + (lane & 15)) * K;
  v8f acc[4];
  v8f z = {0.f, 0.f, 0.f, 0.f, 0.f, 0.f, 0.f, 0.f};
#pragma unroll
  for (int t = 0; t < 4; ++t) acc[t] = z;
  gemm_core4(Abase, Bt, K, c0, lane, acc);

  const int mo = (lane & 16) ? 8 : 0, col = lane & 15;
#pragma unroll
  for (int t = 0; t < 4; ++t) {
    const int c = c0 + t * 16 + col;
    const float bv = bias ? bias[c] : 0.0f;
#pragma unroll
    for (int j = 0; j < 8; ++j) {
      const int r = r0 + j + mo;
      float v = acc[t][j] + bv;
      const size_t idx = (size_t)r * Nc + c;
      if (mode == 1)      outb[idx] = (__bf16)gelu_f(v);
      else if (mode == 2) outf[idx] = res[idx] + v;
      else if (mode == 3) outf[idx] = v;
      else                outb[idx] = (__bf16)v;
    }
  }
}

// ---- QKV GEMM with head-layout scatter epilogue ----------------------------
// q,k -> [B*NH][N][HD] bf16 (unscaled q; SCALE applied on scores)
// v   -> [B*NH][HD][NPAD] bf16 (transposed for the PV B-operand)
__global__ __launch_bounds__(256)
void gemm_qkv_kernel(const __bf16* __restrict__ A, const __bf16* __restrict__ Bt,
                     const float* __restrict__ bias,
                     __bf16* __restrict__ qt, __bf16* __restrict__ kt,
                     __bf16* __restrict__ vt) {
  const int K = CC;
  const int lane = threadIdx.x & 31, wv = threadIdx.x >> 5;
  const int r0 = blockIdx.x * 128 + wv * 16;
  const int c0 = blockIdx.y * 64;
  const __bf16* Abase = A + (size_t)(r0 + (lane & 15)) * K;
  v8f acc[4];
  v8f z = {0.f, 0.f, 0.f, 0.f, 0.f, 0.f, 0.f, 0.f};
#pragma unroll
  for (int t = 0; t < 4; ++t) acc[t] = z;
  gemm_core4(Abase, Bt, K, c0, lane, acc);

  const int mo = (lane & 16) ? 8 : 0, col = lane & 15;
#pragma unroll
  for (int t = 0; t < 4; ++t) {
    const int c = c0 + t * 16 + col;
    const float bv = bias[c];
    const int which = c >> 10, rem = c & 1023, hh = rem >> 6, d = rem & 63;
#pragma unroll
    for (int j = 0; j < 8; ++j) {
      const int m = r0 + j + mo;
      const int b = m / NN, n = m % NN;
      const float v = acc[t][j] + bv;
      const int bh = b * NHEADS + hh;
      if (which == 0)      qt[((size_t)bh * NN + n) * HD + d] = (__bf16)v;
      else if (which == 1) kt[((size_t)bh * NN + n) * HD + d] = (__bf16)v;
      else                 vt[((size_t)bh * HD + d) * NPAD + n] = (__bf16)v;
    }
  }
}

// ---- decomposed rel-pos bias: relh/relw [B*NH][N][28] ----------------------
__global__ void relpos_kernel(const __bf16* __restrict__ qt,
                              const float* __restrict__ rph, const float* __restrict__ rpw,
                              float* __restrict__ relh, float* __restrict__ relw) {
  const int g = blockIdx.x;            // bh*784 + q
  const int q = g % NN;
  const int qh = q / WW_, qw = q % WW_;
  __shared__ float qs[HD];
  const __bf16* qp = qt + (size_t)g * HD;
  if (threadIdx.x < HD) qs[threadIdx.x] = (float)qp[threadIdx.x];
  __syncthreads();
  const int t = threadIdx.x;
  if (t < 28) {
    const float* r = rph + (size_t)(qh - t + 27) * HD;
    float s = 0.f;
#pragma unroll 8
    for (int d = 0; d < HD; ++d) s += qs[d] * r[d];
    relh[(size_t)g * 28 + t] = s;
  } else if (t < 56) {
    const int kw = t - 28;
    const float* r = rpw + (size_t)(qw - kw + 27) * HD;
    float s = 0.f;
#pragma unroll 8
    for (int d = 0; d < HD; ++d) s += qs[d] * r[d];
    relw[(size_t)g * 28 + kw] = s;
  }
}

// ---- fused attention: one block per (b, head, 16-query tile) ---------------
__global__ __launch_bounds__(256)
void attn_kernel(const __bf16* __restrict__ qt, const __bf16* __restrict__ kt,
                 const __bf16* __restrict__ vt,
                 const float* __restrict__ relh, const float* __restrict__ relw,
                 __bf16* __restrict__ attn_out) {
  __shared__ float sc[16][NPAD];     // scores, then unnormalized probs (56 KB)
  __shared__ float red[16][17];
  __shared__ float rowmax[16];
  __shared__ float rowsum[16];
  __shared__ float ob[16][HD];

  const int blk = blockIdx.x;        // bh*49 + qtile
  const int qtile = blk % 49;
  const int bh = blk / 49;
  const int lane = threadIdx.x & 31, wv = threadIdx.x >> 5;

  // q fragments (K = HD = 64 -> two k-steps), unscaled q
  const __bf16* qbase = qt + ((size_t)bh * NN + qtile * 16) * HD;
  const int arow = lane & 15;
  v16bf qf0 = load_fragA(qbase + (size_t)arow * HD + 0, lane);
  v16bf qf1 = load_fragA(qbase + (size_t)arow * HD + 32, lane);

  v8f zv = {0.f, 0.f, 0.f, 0.f, 0.f, 0.f, 0.f, 0.f};

  // ---- scores: QK^T * SCALE + rel_h + rel_w -> LDS -------------------------
  for (int ktile = wv; ktile < 56; ktile += 8) {
    if (ktile < 49) {
      v8f acc = zv;
      const __bf16* kb = kt + ((size_t)bh * NN + ktile * 16 + (lane & 15)) * HD;
      v16bf b0 = load_fragB(kb + 0, lane);
      v16bf b1 = load_fragB(kb + 32, lane);
      acc = __builtin_amdgcn_wmma_f32_16x16x32_bf16(false, qf0, false, b0, (short)0, acc, false, false);
      acc = __builtin_amdgcn_wmma_f32_16x16x32_bf16(false, qf1, false, b1, (short)0, acc, false, false);
      const int mo = (lane & 16) ? 8 : 0, col = lane & 15;
      const int key = ktile * 16 + col, kh = key / WW_, kw = key % WW_;
#pragma unroll
      for (int j = 0; j < 8; ++j) {
        const int q = qtile * 16 + j + mo;
        const size_t rb = (size_t)bh * NN + q;
        sc[j + mo][key] = acc[j] * SCALE_ + relh[rb * 28 + kh] + relw[rb * 28 + kw];
      }
    } else {
      const int col = lane & 15, rbase = (lane & 16) ? 8 : 0;
#pragma unroll
      for (int j = 0; j < 8; ++j) sc[rbase + j][ktile * 16 + col] = -1e30f;
    }
  }
  __syncthreads();

  // ---- softmax (unnormalized probs kept in sc; normalize at the end) -------
  const int row = threadIdx.x >> 4, li = threadIdx.x & 15;
  float mx = -1e30f;
  for (int k2 = li; k2 < NPAD; k2 += 16) mx = fmaxf(mx, sc[row][k2]);
  red[row][li] = mx;
  __syncthreads();
  if (li == 0) {
    float m2 = red[row][0];
#pragma unroll
    for (int i = 1; i < 16; ++i) m2 = fmaxf(m2, red[row][i]);
    rowmax[row] = m2;
  }
  __syncthreads();
  const float mxr = rowmax[row];
  float sum = 0.f;
  for (int k2 = li; k2 < NPAD; k2 += 16) {
    float p = __expf(sc[row][k2] - mxr);
    sc[row][k2] = p;
    sum += p;
  }
  red[row][li] = sum;
  __syncthreads();
  if (li == 0) {
    float s2 = 0.f;
#pragma unroll
    for (int i = 0; i < 16; ++i) s2 += red[row][i];
    rowsum[row] = s2;
  }
  for (int i = threadIdx.x; i < 16 * HD; i += 256) ((float*)ob)[i] = 0.f;
  __syncthreads();

  // ---- PV: probs[16][896] @ v[896][64]; 8 waves = 4 dim-tiles x 2 key halves
  const int dt = wv & 3, khalf = wv >> 2;
  v8f acc = zv;
  const int qrow = lane & 15;
  const int kbA = (lane & 16) ? 8 : 0;
  for (int s = 0; s < 14; ++s) {
    const int kk = khalf * 448 + s * 32;
    v16bf a;
#pragma unroll
    for (int e = 0; e < 8; ++e) {
      a[e]     = (__bf16)sc[qrow][kk + kbA + e];
      a[8 + e] = (__bf16)sc[qrow][kk + kbA + 16 + e];
    }
    const __bf16* vb = vt + ((size_t)bh * HD + dt * 16 + (lane & 15)) * NPAD + kk;
    v16bf b = load_fragB(vb, lane);
    acc = __builtin_amdgcn_wmma_f32_16x16x32_bf16(false, a, false, b, (short)0, acc, false, false);
  }
  {
    const int mo = (lane & 16) ? 8 : 0, col = lane & 15;
#pragma unroll
    for (int j = 0; j < 8; ++j) atomicAdd(&ob[j + mo][dt * 16 + col], acc[j]);
  }
  __syncthreads();

  // ---- write out (B, N, C) bf16 -------------------------------------------
  const int b_ = bh >> 4, h_ = bh & 15;
  for (int i = threadIdx.x; i < 16 * HD; i += 256) {
    const int qr = i >> 6, d = i & 63;
    const int n = qtile * 16 + qr;
    const float val = ob[qr][d] / rowsum[qr];
    attn_out[((size_t)b_ * NN + n) * CC + h_ * HD + d] = (__bf16)val;
  }
}

// ---- 3x3 SAME conv as implicit GEMM (K = 9*512) ----------------------------
__global__ __launch_bounds__(256)
void conv3x3_kernel(const __bf16* __restrict__ in /*[M][512]*/,
                    const __bf16* __restrict__ wt /*[9][co=512][ci=512]*/,
                    float* __restrict__ out /*[M][512]*/) {
  const int lane = threadIdx.x & 31, wv = threadIdx.x >> 5;
  const int r0 = blockIdx.x * 128 + wv * 16;
  const int c0 = blockIdx.y * 64;
  const int m = r0 + (lane & 15);
  const int bimg = m / NN, p = m % NN, hh = p / WW_, ww = p % WW_;
  v8f acc[4];
  v8f z = {0.f, 0.f, 0.f, 0.f, 0.f, 0.f, 0.f, 0.f};
#pragma unroll
  for (int t = 0; t < 4; ++t) acc[t] = z;

  for (int tap = 0; tap < 9; ++tap) {
    const int dy = tap / 3 - 1, dx = tap % 3 - 1;
    const int nh = hh + dy, nw = ww + dx;
    const bool valid = (nh >= 0) && (nh < HH_) && (nw >= 0) && (nw < WW_);
    const __bf16* abase = in + (size_t)(bimg * NN + nh * WW_ + nw) * BC_;
    const __bf16* wbase = wt + (size_t)tap * BC_ * BC_;
    for (int kk = 0; kk < BC_; kk += 32) {
      v16bf a = load_fragA_guard(abase + kk, lane, valid);
#pragma unroll
      for (int t = 0; t < 4; ++t) {
        v16bf b = load_fragB(wbase + (size_t)(c0 + t * 16 + (lane & 15)) * BC_ + kk, lane);
        acc[t] = __builtin_amdgcn_wmma_f32_16x16x32_bf16(
            false, a, false, b, (short)0, acc[t], false, false);
      }
    }
  }
  const int mo = (lane & 16) ? 8 : 0, col = lane & 15;
#pragma unroll
  for (int t = 0; t < 4; ++t) {
    const int c = c0 + t * 16 + col;
#pragma unroll
    for (int j = 0; j < 8; ++j)
      out[(size_t)(r0 + j + mo) * BC_ + c] = acc[t][j];
  }
}

// ---- LayerNorm; mode 0: ->bf16 | 1: GELU(LN)->bf16 | 2: res + LN -> f32 ----
__global__ void ln_kernel(const float* __restrict__ in, const float* __restrict__ w,
                          const float* __restrict__ b, int Crow, float eps, int mode,
                          __bf16* __restrict__ outb, float* __restrict__ outf,
                          const float* __restrict__ res) {
  const int row = blockIdx.x;
  const float* x = in + (size_t)row * Crow;
  __shared__ float red[256];
  float s = 0.f;
  for (int i = threadIdx.x; i < Crow; i += 256) s += x[i];
  red[threadIdx.x] = s;
  __syncthreads();
  for (int o = 128; o > 0; o >>= 1) {
    if (threadIdx.x < o) red[threadIdx.x] += red[threadIdx.x + o];
    __syncthreads();
  }
  const float mean = red[0] / (float)Crow;
  __syncthreads();
  float v = 0.f;
  for (int i = threadIdx.x; i < Crow; i += 256) {
    const float d = x[i] - mean;
    v += d * d;
  }
  red[threadIdx.x] = v;
  __syncthreads();
  for (int o = 128; o > 0; o >>= 1) {
    if (threadIdx.x < o) red[threadIdx.x] += red[threadIdx.x + o];
    __syncthreads();
  }
  const float rstd = rsqrtf(red[0] / (float)Crow + eps);
  for (int i = threadIdx.x; i < Crow; i += 256) {
    const float val = (x[i] - mean) * rstd * w[i] + b[i];
    const size_t idx = (size_t)row * Crow + i;
    if (mode == 0)      outb[idx] = (__bf16)val;
    else if (mode == 1) outb[idx] = (__bf16)gelu_f(val);
    else                outf[idx] = res[idx] + val;
  }
}

// ---- weight transpose fp32[b][r][c] -> bf16[b][c][r] -----------------------
__global__ void transpose_w_kernel(const float* __restrict__ in, __bf16* __restrict__ out,
                                   int R, int Ccols, size_t total) {
  const size_t idx = (size_t)blockIdx.x * 256 + threadIdx.x;
  if (idx >= total) return;
  const size_t rc = (size_t)R * Ccols;
  const size_t bidx = idx / rc;
  const size_t rem = idx % rc;
  const int r = (int)(rem / Ccols), c = (int)(rem % Ccols);
  out[bidx * rc + (size_t)c * R + r] = (__bf16)in[idx];
}

__global__ void to_bf16_kernel(const float* __restrict__ in, __bf16* __restrict__ out,
                               size_t n) {
  const size_t i = (size_t)blockIdx.x * 256 + threadIdx.x;
  if (i < n) out[i] = (__bf16)in[i];
}

// ---------------------------------------------------------------------------
extern "C" void kernel_launch(void* const* d_in, const int* in_sizes, int n_in,
                              void* d_out, int out_size, void* d_ws, size_t ws_size,
                              hipStream_t stream) {
  (void)in_sizes; (void)n_in; (void)out_size; (void)ws_size;
  const float* x     = (const float*)d_in[0];
  const float* ln1w  = (const float*)d_in[1];
  const float* ln1b  = (const float*)d_in[2];
  const float* qkvW  = (const float*)d_in[3];
  const float* qkvB  = (const float*)d_in[4];
  const float* projW = (const float*)d_in[5];
  const float* projB = (const float*)d_in[6];
  const float* rph   = (const float*)d_in[7];
  const float* rpw   = (const float*)d_in[8];
  const float* ln2w  = (const float*)d_in[9];
  const float* ln2b  = (const float*)d_in[10];
  const float* w1    = (const float*)d_in[11];
  const float* b1    = (const float*)d_in[12];
  const float* w2    = (const float*)d_in[13];
  const float* b2    = (const float*)d_in[14];
  const float* c1w   = (const float*)d_in[15];
  const float* cl1w  = (const float*)d_in[16];
  const float* cl1b  = (const float*)d_in[17];
  const float* c2w   = (const float*)d_in[18];
  const float* cl2w  = (const float*)d_in[19];
  const float* cl2b  = (const float*)d_in[20];
  const float* c3w   = (const float*)d_in[21];
  const float* cl3w  = (const float*)d_in[22];
  const float* cl3b  = (const float*)d_in[23];
  float* out = (float*)d_out;

  char* ws = (char*)d_ws;
  size_t off = 0;
  auto alloc = [&](size_t bytes) -> void* {
    void* p = ws + off;
    off += (bytes + 255) & ~(size_t)255;
    return p;
  };
  __bf16* h_b   = (__bf16*)alloc((size_t)MM * CC * 2);
  __bf16* qkvt  = (__bf16*)alloc((size_t)3 * CC * CC * 2);
  __bf16* projt = (__bf16*)alloc((size_t)CC * CC * 2);
  __bf16* w1t   = (__bf16*)alloc((size_t)HID_ * CC * 2);
  __bf16* w2t   = (__bf16*)alloc((size_t)CC * HID_ * 2);
  __bf16* c1t   = (__bf16*)alloc((size_t)BC_ * CC * 2);
  __bf16* c2t   = (__bf16*)alloc((size_t)9 * BC_ * BC_ * 2);
  __bf16* c3t   = (__bf16*)alloc((size_t)CC * BC_ * 2);
  __bf16* qt    = (__bf16*)alloc((size_t)BB * NHEADS * NN * HD * 2);
  __bf16* kt    = (__bf16*)alloc((size_t)BB * NHEADS * NN * HD * 2);
  size_t vt_bytes = (size_t)BB * NHEADS * HD * NPAD * 2;
  __bf16* vt    = (__bf16*)alloc(vt_bytes);
  float*  relh  = (float*)alloc((size_t)BB * NHEADS * NN * 28 * 4);
  float*  relw  = (float*)alloc((size_t)BB * NHEADS * NN * 28 * 4);
  __bf16* attnb = (__bf16*)alloc((size_t)MM * CC * 2);
  float*  xt    = (float*)alloc((size_t)MM * CC * 4);
  __bf16* h2b   = (__bf16*)alloc((size_t)MM * CC * 2);
  __bf16* hidb  = (__bf16*)alloc((size_t)MM * HID_ * 2);
  float*  y     = (float*)alloc((size_t)MM * CC * 4);
  __bf16* yb    = (__bf16*)alloc((size_t)MM * CC * 2);
  float*  o1    = (float*)alloc((size_t)MM * BC_ * 4);
  __bf16* o1b   = (__bf16*)alloc((size_t)MM * BC_ * 2);
  float*  o2    = (float*)alloc((size_t)MM * BC_ * 4);
  __bf16* o2b   = (__bf16*)alloc((size_t)MM * BC_ * 2);
  float*  o3    = (float*)alloc((size_t)MM * CC * 4);

  // -- weight transposes (fp32 [in][out] -> bf16 [out][in]) ------------------
  auto tgrid = [](size_t tot) { return (unsigned)((tot + 255) / 256); };
  {
    size_t t;
    t = (size_t)CC * 3 * CC;   transpose_w_kernel<<<tgrid(t), 256, 0, stream>>>(qkvW, qkvt, CC, 3 * CC, t);
    t = (size_t)CC * CC;       transpose_w_kernel<<<tgrid(t), 256, 0, stream>>>(projW, projt, CC, CC, t);
    t = (size_t)CC * HID_;     transpose_w_kernel<<<tgrid(t), 256, 0, stream>>>(w1, w1t, CC, HID_, t);
    t = (size_t)HID_ * CC;     transpose_w_kernel<<<tgrid(t), 256, 0, stream>>>(w2, w2t, HID_, CC, t);
    t = (size_t)CC * BC_;      transpose_w_kernel<<<tgrid(t), 256, 0, stream>>>(c1w, c1t, CC, BC_, t);
    t = (size_t)9 * BC_ * BC_; transpose_w_kernel<<<tgrid(t), 256, 0, stream>>>(c2w, c2t, BC_, BC_, t);
    t = (size_t)BC_ * CC;      transpose_w_kernel<<<tgrid(t), 256, 0, stream>>>(c3w, c3t, BC_, CC, t);
  }
  hipMemsetAsync(vt, 0, vt_bytes, stream);  // zero key-padding for PV

  // -- LN1 -> bf16 -----------------------------------------------------------
  ln_kernel<<<MM, 256, 0, stream>>>(x, ln1w, ln1b, CC, EPS_LN, 0, h_b, nullptr, nullptr);
  // -- QKV GEMM (scatter to head layouts) ------------------------------------
  gemm_qkv_kernel<<<dim3(MM / 128, (3 * CC) / 64), 256, 0, stream>>>(h_b, qkvt, qkvB, qt, kt, vt);
  // -- rel-pos bias tables ---------------------------------------------------
  relpos_kernel<<<BB * NHEADS * NN, 64, 0, stream>>>(qt, rph, rpw, relh, relw);
  // -- fused attention -------------------------------------------------------
  attn_kernel<<<BB * NHEADS * 49, 256, 0, stream>>>(qt, kt, vt, relh, relw, attnb);
  // -- proj GEMM + residual(x) -> xt -----------------------------------------
  gemm_ws_kernel<<<dim3(MM / 128, CC / 64), 256, 0, stream>>>(attnb, projt, projB, x, xt, nullptr, CC, CC, 2);
  // -- LN2 -> bf16 -----------------------------------------------------------
  ln_kernel<<<MM, 256, 0, stream>>>(xt, ln2w, ln2b, CC, EPS_LN, 0, h2b, nullptr, nullptr);
  // -- MLP1 (+GELU) ----------------------------------------------------------
  gemm_ws_kernel<<<dim3(MM / 128, HID_ / 64), 256, 0, stream>>>(h2b, w1t, b1, nullptr, nullptr, hidb, HID_, CC, 1);
  // -- MLP2 + residual(xt) -> y ----------------------------------------------
  gemm_ws_kernel<<<dim3(MM / 128, CC / 64), 256, 0, stream>>>(hidb, w2t, b2, xt, y, nullptr, CC, HID_, 2);
  // -- conv branch -----------------------------------------------------------
  to_bf16_kernel<<<tgrid((size_t)MM * CC), 256, 0, stream>>>(y, yb, (size_t)MM * CC);
  gemm_ws_kernel<<<dim3(MM / 128, BC_ / 64), 256, 0, stream>>>(yb, c1t, nullptr, nullptr, o1, nullptr, BC_, CC, 3);
  ln_kernel<<<MM, 256, 0, stream>>>(o1, cl1w, cl1b, BC_, EPS_CLN, 1, o1b, nullptr, nullptr);
  conv3x3_kernel<<<dim3(MM / 128, BC_ / 64), 256, 0, stream>>>(o1b, c2t, o2);
  ln_kernel<<<MM, 256, 0, stream>>>(o2, cl2w, cl2b, BC_, EPS_CLN, 1, o2b, nullptr, nullptr);
  gemm_ws_kernel<<<dim3(MM / 128, CC / 64), 256, 0, stream>>>(o2b, c3t, nullptr, nullptr, o3, nullptr, CC, BC_, 3);
  // -- final: out = y + LN(o3) ----------------------------------------------
  ln_kernel<<<MM, 256, 0, stream>>>(o3, cl3w, cl3b, CC, EPS_CLN, 2, nullptr, out, y);
}
